// ALCDEFTemporalGNN_79963701117497
// MI455X (gfx1250) — compile-verified, matrix operand order checked by reading, not run on hardware
//
#include <hip/hip_runtime.h>
#include <hip/hip_bf16.h>

#define H 128
#define TSTEPS 200

typedef __attribute__((ext_vector_type(16))) _Float16 v16h;
typedef __attribute__((ext_vector_type(8)))  _Float16 v8h;
typedef __attribute__((ext_vector_type(8)))  float    v8f;

static __device__ __forceinline__ v8f v8f_zero() {
  v8f z;
#pragma unroll
  for (int i = 0; i < 8; ++i) z[i] = 0.f;
  return z;
}

// Opaque zero offset: defeats LICM on loads whose address includes it (the
// weight stream must be re-read each timestep, not hoisted+spilled), while
// keeping pointer provenance (kernarg -> global address space -> global_load).
static __device__ __forceinline__ int launder_zero() {
  int z = 0;
#if defined(__HIP_DEVICE_COMPILE__)
  asm volatile("" : "+s"(z));
#endif
  return z;
}

static __device__ __forceinline__ v8f wmma16(v16h a, v16h b, v8f c) {
  // D = A(16x32 f16) * B(32x16 f16) + C(16x16 f32)
  return __builtin_amdgcn_wmma_f32_16x16x32_f16(
      false, a, false, b, (short)0, c, false, false);
}

// Build a 16x32 f16 A-fragment from an LDS row-major [rows][H] f16 buffer.
// ISA layout (16-bit A 16x32): lanes 0-15 -> M=lane, K pattern {0..7,16..23};
// lanes 16-31 -> M=lane-16, K pattern {8..15,24..31}.
static __device__ __forceinline__ v16h afrag_lds(const _Float16* buf, int r0,
                                                 int kk, int lane) {
  int row   = r0 + (lane & 15);
  int kbase = kk * 32 + ((lane & 16) ? 8 : 0);
  v8h lo = *(const v8h*)(buf + row * H + kbase);
  v8h hi = *(const v8h*)(buf + row * H + kbase + 16);
  v16h a;
#pragma unroll
  for (int p = 0; p < 8; ++p) { a[p] = lo[p]; a[p + 8] = hi[p]; }
  return a;
}

// Pre-packed B fragment: blob = 512 halves, lane-major (16 contiguous per lane).
static __device__ __forceinline__ v16h bfrag(const _Float16* bp, int blob,
                                             int lane) {
  return *(const v16h*)(bp + (size_t)blob * 512 + lane * 16);
}

static __device__ __forceinline__ float sigf(float v) {
  return 1.f / (1.f + __expf(-v));
}

// ---------------------------------------------------------------------------
// Weight packing: produce per-fragment layout matching the B operand of
// v_wmma_f32_16x16x32_f16. Output half idx -> (blob, lane, p):
//   k = kk*32 + (p<8?0:16) + (lane>=16?8:0) + (p&7),  n = j*16 + (lane&15)
// mode 0: B[k][n] = src0[n*K + k]            (gates = h @ W^T, W is [N][K])
// mode 1: B[k][n] = k<128 ? src0[n*128+k] : src1[n*128+(k-128)]  (layer1 fused)
// mode 2: B[k][n] = src0[k*Ntot + n]         (m = h @ Wg, Wg row-major [K][N])
// ---------------------------------------------------------------------------
__global__ void pack_b_kernel(const float* __restrict__ src0,
                              const float* __restrict__ src1,
                              _Float16* __restrict__ outp, int K, int Ntot,
                              int mode) {
  int idx = blockIdx.x * blockDim.x + threadIdx.x;
  int total = K * Ntot;
  if (idx >= total) return;
  int p = idx & 15;
  int lane = (idx >> 4) & 31;
  int blob = idx >> 9;
  int Ntiles = Ntot >> 4;
  int kk = blob / Ntiles;
  int j  = blob % Ntiles;
  int k = kk * 32 + ((p < 8) ? 0 : 16) + ((lane >= 16) ? 8 : 0) + (p & 7);
  int n = j * 16 + (lane & 15);
  float v;
  if (mode == 0)      v = src0[(size_t)n * K + k];
  else if (mode == 1) v = (k < 128) ? src0[(size_t)n * 128 + k]
                                    : src1[(size_t)n * 128 + (k - 128)];
  else                v = src0[(size_t)k * Ntot + n];
  outp[idx] = (_Float16)v;
}

__global__ void bias_sum_kernel(const float* bi0, const float* bh0,
                                const float* bi1, const float* bh1,
                                float* b0s, float* b1s) {
  int i = blockIdx.x * blockDim.x + threadIdx.x;
  if (i < 512) { b0s[i] = bi0[i] + bh0[i]; b1s[i] = bi1[i] + bh1[i]; }
}

// ---------------------------------------------------------------------------
// Persistent fused 2-layer LSTM. Block = 4 waves = 64 nodes. Each wave owns a
// 16-node strip: c-state lives in f32 WMMA C-layout accumulators, h-state is
// staged f16 in (wave-private rows of) LDS to rebuild A-fragments each step.
// Gate GEMMs: layer0 [16x128]x[128x512]; layer1 [16x256]x[256x512] (ih|hh).
// Weights stream from global each timestep (L2-resident); x_t staging uses the
// CDNA5 async global->LDS path (ASYNCcnt).
// ---------------------------------------------------------------------------
__global__ __launch_bounds__(128, 1)
void lstm2_kernel(const float* __restrict__ x,        // [N][T]
                  const _Float16* __restrict__ BP0g,  // packed, 4x32 blobs
                  const _Float16* __restrict__ BP1g,  // packed, 8x32 blobs
                  const float* __restrict__ wih0g,    // [512]
                  const float* __restrict__ b0g,      // [512]
                  const float* __restrict__ b1g,      // [512]
                  float* __restrict__ hout,           // [N][H]
                  int nNodes) {
  __shared__ __align__(16) _Float16 h1buf[64 * H];
  __shared__ __align__(16) _Float16 h2buf[64 * H];
  __shared__ float xbuf[64];
  const int tid = threadIdx.x, wave = tid >> 5, lane = tid & 31;
  const int nodeBase = blockIdx.x * 64;
  const int r0 = wave * 16;
  const int cn = lane & 15;         // column within 16-tile (C layout)
  const int mr = (lane & 16) >> 1;  // row offset 0/8 (C layout)

  for (int i = tid; i < 64 * H; i += 128) {
    h1buf[i] = (_Float16)0.f;
    h2buf[i] = (_Float16)0.f;
  }
  v8f c1[8], c2[8];
#pragma unroll
  for (int j = 0; j < 8; ++j) { c1[j] = v8f_zero(); c2[j] = v8f_zero(); }
  __syncthreads();

  for (int t = 0; t < TSTEPS; ++t) {
    // Per-iteration opaque offset: keeps weight loads inside the loop (no
    // hoist/spill) while preserving global address-space provenance.
    const int zo = launder_zero();
    const _Float16* BP0 = BP0g + zo;
    const _Float16* BP1 = BP1g + zo;
    const float* b0   = b0g + zo;
    const float* b1   = b1g + zo;
    const float* wih0 = wih0g + zo;

    // Stage x_t via CDNA5 async global->LDS copy (tracked by ASYNCcnt).
    if (tid < 64) {
      int node = nodeBase + tid;
      if (node < nNodes) {
#if defined(__HIP_DEVICE_COMPILE__)
        unsigned ldsa = (unsigned)(size_t)&xbuf[tid];
        const float* gp = x + (size_t)node * TSTEPS + t;
        asm volatile("global_load_async_to_lds_b32 %0, %1, off"
                     :: "v"(ldsa), "v"(gp) : "memory");
#else
        xbuf[tid] = x[(size_t)node * TSTEPS + t];
#endif
      } else {
        xbuf[tid] = 0.f;
      }
    }
#if defined(__HIP_DEVICE_COMPILE__)
    asm volatile("s_wait_asynccnt 0x0" ::: "memory");
#endif
    __syncthreads();

    // ---------------- layer 0: gates = h1 @ W_hh0^T + x_t*w_ih0 + b ----------
    v16h a0[4];
#pragma unroll
    for (int kk = 0; kk < 4; ++kk) a0[kk] = afrag_lds(h1buf, r0, kk, lane);
#pragma unroll
    for (int j = 0; j < 8; ++j) {
      __builtin_prefetch(BP0 + (size_t)(((j + 1) & 7)) * 512, 0, 1);
      v8f ai = v8f_zero(), af = v8f_zero(), ag = v8f_zero(), ao = v8f_zero();
#pragma unroll
      for (int kk = 0; kk < 4; ++kk) {
        ai = wmma16(a0[kk], bfrag(BP0, kk * 32 + 0 * 8 + j, lane), ai);
        af = wmma16(a0[kk], bfrag(BP0, kk * 32 + 1 * 8 + j, lane), af);
        ag = wmma16(a0[kk], bfrag(BP0, kk * 32 + 2 * 8 + j, lane), ag);
        ao = wmma16(a0[kk], bfrag(BP0, kk * 32 + 3 * 8 + j, lane), ao);
      }
      const int col = j * 16 + cn;
      const float bi = b0[col], bf = b0[128 + col], bg_ = b0[256 + col],
                  bo = b0[384 + col];
      const float wi = wih0[col], wf = wih0[128 + col], wg = wih0[256 + col],
                  wo = wih0[384 + col];
#pragma unroll
      for (int v = 0; v < 8; ++v) {
        const float xv = xbuf[r0 + mr + v];
        const float gi = sigf(ai[v] + bi + xv * wi);
        const float gf = sigf(af[v] + bf + xv * wf);
        const float gg = tanhf(ag[v] + bg_ + xv * wg);
        const float go = sigf(ao[v] + bo + xv * wo);
        const float c = gf * c1[j][v] + gi * gg;
        c1[j][v] = c;
        h1buf[(r0 + mr + v) * H + col] = (_Float16)(go * tanhf(c));
      }
    }

    // ------------- layer 1: gates = [h1new|h2] @ [W_ih1|W_hh1]^T + b --------
    v16h a1[8];
#pragma unroll
    for (int kk = 0; kk < 4; ++kk) {
      a1[kk]     = afrag_lds(h1buf, r0, kk, lane);
      a1[4 + kk] = afrag_lds(h2buf, r0, kk, lane);
    }
#pragma unroll
    for (int j = 0; j < 8; ++j) {
      __builtin_prefetch(BP1 + (size_t)(((j + 1) & 7)) * 512, 0, 1);
      v8f ai = v8f_zero(), af = v8f_zero(), ag = v8f_zero(), ao = v8f_zero();
#pragma unroll
      for (int kk = 0; kk < 8; ++kk) {
        ai = wmma16(a1[kk], bfrag(BP1, kk * 32 + 0 * 8 + j, lane), ai);
        af = wmma16(a1[kk], bfrag(BP1, kk * 32 + 1 * 8 + j, lane), af);
        ag = wmma16(a1[kk], bfrag(BP1, kk * 32 + 2 * 8 + j, lane), ag);
        ao = wmma16(a1[kk], bfrag(BP1, kk * 32 + 3 * 8 + j, lane), ao);
      }
      const int col = j * 16 + cn;
      const float bi = b1[col], bf = b1[128 + col], bg_ = b1[256 + col],
                  bo = b1[384 + col];
#pragma unroll
      for (int v = 0; v < 8; ++v) {
        const float gi = sigf(ai[v] + bi);
        const float gf = sigf(af[v] + bf);
        const float gg = tanhf(ag[v] + bg_);
        const float go = sigf(ao[v] + bo);
        const float c = gf * c2[j][v] + gi * gg;
        c2[j][v] = c;
        const float hn = go * tanhf(c);
        h2buf[(r0 + mr + v) * H + col] = (_Float16)hn;
        if (t == TSTEPS - 1) {
          const int node = nodeBase + r0 + mr + v;
          if (node < nNodes) hout[(size_t)node * H + col] = hn;
        }
      }
    }
    __syncthreads();
  }
}

// ---------------------------------------------------------------------------
// GCN dense part: m = h @ Wg (f16 WMMA, f32 accumulate). Block = 8 waves,
// 128-row tile staged as f16 in LDS.
// ---------------------------------------------------------------------------
__global__ __launch_bounds__(256)
void gcn_gemm_kernel(const float* __restrict__ hin,
                     const _Float16* __restrict__ BPg,
                     float* __restrict__ mout, int nNodes) {
  __shared__ __align__(16) _Float16 hbuf[128 * H];
  const int tid = threadIdx.x, wave = tid >> 5, lane = tid & 31;
  const long nodeBase = (long)blockIdx.x * 128;
  for (int i = tid; i < 128 * H; i += 256) {
    int r = i >> 7, c = i & 127;
    long node = nodeBase + r;
    hbuf[i] = (_Float16)((node < nNodes) ? hin[node * H + c] : 0.f);
  }
  __syncthreads();
  const int r0 = wave * 16, cn = lane & 15, mr = (lane & 16) >> 1;
  v16h a[4];
#pragma unroll
  for (int kk = 0; kk < 4; ++kk) a[kk] = afrag_lds(hbuf, r0, kk, lane);
#pragma unroll
  for (int j = 0; j < 8; ++j) {
    v8f acc = v8f_zero();
#pragma unroll
    for (int kk = 0; kk < 4; ++kk)
      acc = wmma16(a[kk], bfrag(BPg, kk * 8 + j, lane), acc);
#pragma unroll
    for (int v = 0; v < 8; ++v) {
      long node = nodeBase + r0 + mr + v;
      if (node < nNodes) mout[node * H + j * 16 + cn] = acc[v];
    }
  }
}

__global__ void deg_init_kernel(float* deg, int n) {
  int i = blockIdx.x * blockDim.x + threadIdx.x;
  if (i < n) deg[i] = 1.0f;  // self loop
}
__global__ void deg_accum_kernel(const int* __restrict__ col, float* deg,
                                 long e) {
  long i = (long)blockIdx.x * blockDim.x + threadIdx.x;
  if (i < e) atomicAdd(&deg[col[i]], 1.0f);
}
__global__ void deg_finish_kernel(float* deg, int n) {
  int i = blockIdx.x * blockDim.x + threadIdx.x;
  if (i < n) deg[i] = rsqrtf(fmaxf(deg[i], 1.0f));  // in-place -> dinv
}
__global__ void zero_kernel(float* p, long n) {
  long i = (long)blockIdx.x * blockDim.x + threadIdx.x;
  if (i < n) p[i] = 0.f;
}

// One wave per edge; each lane scatters 4 dims via float atomics.
__global__ __launch_bounds__(256)
void edge_scatter_kernel(const int* __restrict__ row,
                         const int* __restrict__ col,
                         const float* __restrict__ m,
                         const float* __restrict__ dinv,
                         float* __restrict__ outp, long nEdges) {
  long gw = ((long)blockIdx.x * blockDim.x + threadIdx.x) >> 5;
  int lane = threadIdx.x & 31;
  if (gw >= nEdges) return;
  int r = row[gw], c = col[gw];
  float w = dinv[r] * dinv[c];
  float4 mv = ((const float4*)(m + (size_t)r * H))[lane];
  float* oc = outp + (size_t)c * H + lane * 4;
  atomicAdd(oc + 0, mv.x * w);
  atomicAdd(oc + 1, mv.y * w);
  atomicAdd(oc + 2, mv.z * w);
  atomicAdd(oc + 3, mv.w * w);
}

__global__ void self_bias_kernel(const float* __restrict__ m,
                                 const float* __restrict__ dinv,
                                 const float* __restrict__ bg,
                                 float* __restrict__ outp, long total) {
  long i = (long)blockIdx.x * blockDim.x + threadIdx.x;
  if (i >= total) return;
  long n = i >> 7;
  int d = i & 127;
  float di = dinv[n];
  outp[i] += m[i] * di * di + bg[d];
}

// LayerNorm + ReLU + residual, one wave per node (4 dims/lane).
__global__ __launch_bounds__(256)
void ln_res_kernel(const float* __restrict__ agg,
                   const float* __restrict__ gamma,
                   const float* __restrict__ beta,
                   float* __restrict__ h, long nNodes) {
  long gw = ((long)blockIdx.x * blockDim.x + threadIdx.x) >> 5;
  int lane = threadIdx.x & 31;
  if (gw >= nNodes) return;
  float4 v = ((const float4*)(agg + gw * H))[lane];
  float s = v.x + v.y + v.z + v.w;
#pragma unroll
  for (int m = 16; m >= 1; m >>= 1) s += __shfl_xor(s, m, 32);
  const float mu = s * (1.0f / H);
  float dx0 = v.x - mu, dx1 = v.y - mu, dx2 = v.z - mu, dx3 = v.w - mu;
  float q = dx0 * dx0 + dx1 * dx1 + dx2 * dx2 + dx3 * dx3;
#pragma unroll
  for (int m = 16; m >= 1; m >>= 1) q += __shfl_xor(q, m, 32);
  const float inv = rsqrtf(q * (1.0f / H) + 1e-5f);
  float4 g4 = ((const float4*)gamma)[lane];
  float4 b4 = ((const float4*)beta)[lane];
  float4 hp = ((const float4*)(h + gw * H))[lane];
  float4 o;
  o.x = fmaxf(dx0 * inv * g4.x + b4.x, 0.f) + hp.x;
  o.y = fmaxf(dx1 * inv * g4.y + b4.y, 0.f) + hp.y;
  o.z = fmaxf(dx2 * inv * g4.z + b4.z, 0.f) + hp.z;
  o.w = fmaxf(dx3 * inv * g4.w + b4.w, 0.f) + hp.w;
  ((float4*)(h + gw * H))[lane] = o;
}

__global__ __launch_bounds__(256)
void pool_accum_kernel(const float* __restrict__ h, float* __restrict__ pooled,
                       int n) {
  int d = threadIdx.x & 127;
  int sub = threadIdx.x >> 7;  // 0..1
  long r0 = (long)blockIdx.x * 64;
  float s = 0.f;
  for (int r = sub; r < 64; r += 2) {
    long node = r0 + r;
    if (node < n) s += h[node * H + d];
  }
  atomicAdd(&pooled[d], s);
}

__global__ void head_kernel(const float* __restrict__ pooled,
                            const float* __restrict__ Wout,
                            const float* __restrict__ bout,
                            float* __restrict__ outp, int nNodes) {
  int lane = threadIdx.x;
  float s = 0.f;
  for (int d = lane; d < H; d += 32) s += pooled[d] * Wout[d];
#pragma unroll
  for (int m = 16; m >= 1; m >>= 1) s += __shfl_xor(s, m, 32);
  if (lane == 0) outp[0] = s / (float)nNodes + bout[0];
}

extern "C" void kernel_launch(void* const* d_in, const int* in_sizes, int n_in,
                              void* d_out, int out_size, void* d_ws,
                              size_t ws_size, hipStream_t stream) {
  (void)n_in; (void)out_size; (void)ws_size;
  const float* x     = (const float*)d_in[0];
  const int*   ei    = (const int*)d_in[1];
  const float* W_ih0 = (const float*)d_in[2];
  const float* W_hh0 = (const float*)d_in[3];
  const float* b_ih0 = (const float*)d_in[4];
  const float* b_hh0 = (const float*)d_in[5];
  const float* W_ih1 = (const float*)d_in[6];
  const float* W_hh1 = (const float*)d_in[7];
  const float* b_ih1 = (const float*)d_in[8];
  const float* b_hh1 = (const float*)d_in[9];
  const float* Wg[3] = {(const float*)d_in[10], (const float*)d_in[14],
                        (const float*)d_in[18]};
  const float* bg[3] = {(const float*)d_in[11], (const float*)d_in[15],
                        (const float*)d_in[19]};
  const float* gm[3] = {(const float*)d_in[12], (const float*)d_in[16],
                        (const float*)d_in[20]};
  const float* bt[3] = {(const float*)d_in[13], (const float*)d_in[17],
                        (const float*)d_in[21]};
  const float* W_out = (const float*)d_in[22];
  const float* b_out = (const float*)d_in[23];

  const int  nNodes = in_sizes[0] / TSTEPS;
  const long nEdges = (long)in_sizes[1] / 2;
  const int* erow = ei;
  const int* ecol = ei + nEdges;

  char* base = (char*)d_ws;
  size_t off = 0;
  auto alloc = [&](size_t bytes) -> void* {
    void* p = base + off;
    off = (off + bytes + 255) & ~(size_t)255;
    return p;
  };
  _Float16* BP0  = (_Float16*)alloc((size_t)128 * 512 * 2);
  _Float16* BP1  = (_Float16*)alloc((size_t)256 * 512 * 2);
  _Float16* BPg0 = (_Float16*)alloc((size_t)128 * 128 * 2);
  _Float16* BPg1 = (_Float16*)alloc((size_t)128 * 128 * 2);
  _Float16* BPg2 = (_Float16*)alloc((size_t)128 * 128 * 2);
  float* b0s    = (float*)alloc(512 * 4);
  float* b1s    = (float*)alloc(512 * 4);
  float* dinv   = (float*)alloc((size_t)nNodes * 4);
  float* hA     = (float*)alloc((size_t)nNodes * H * 4);
  float* hB     = (float*)alloc((size_t)nNodes * H * 4);
  float* hC     = (float*)alloc((size_t)nNodes * H * 4);
  float* pooled = (float*)alloc(128 * 4);

  // Weight packing + bias fusion (tiny, L2-resident afterwards).
  pack_b_kernel<<<(128 * 512 + 255) / 256, 256, 0, stream>>>(W_hh0, nullptr,
                                                             BP0, 128, 512, 0);
  pack_b_kernel<<<(256 * 512 + 255) / 256, 256, 0, stream>>>(W_ih1, W_hh1,
                                                             BP1, 256, 512, 1);
  pack_b_kernel<<<(128 * 128 + 255) / 256, 256, 0, stream>>>(Wg[0], nullptr,
                                                             BPg0, 128, 128, 2);
  pack_b_kernel<<<(128 * 128 + 255) / 256, 256, 0, stream>>>(Wg[1], nullptr,
                                                             BPg1, 128, 128, 2);
  pack_b_kernel<<<(128 * 128 + 255) / 256, 256, 0, stream>>>(Wg[2], nullptr,
                                                             BPg2, 128, 128, 2);
  bias_sum_kernel<<<2, 256, 0, stream>>>(b_ih0, b_hh0, b_ih1, b_hh1, b0s, b1s);

  // Persistent fused 2-layer LSTM -> hA = final h2 [N][H].
  lstm2_kernel<<<(nNodes + 63) / 64, 128, 0, stream>>>(x, BP0, BP1, W_ih0, b0s,
                                                       b1s, hA, nNodes);

  // GCN symmetric-norm degree terms.
  deg_init_kernel<<<(nNodes + 255) / 256, 256, 0, stream>>>(dinv, nNodes);
  deg_accum_kernel<<<(int)((nEdges + 255) / 256), 256, 0, stream>>>(ecol, dinv,
                                                                    nEdges);
  deg_finish_kernel<<<(nNodes + 255) / 256, 256, 0, stream>>>(dinv, nNodes);

  const _Float16* BPgl[3] = {BPg0, BPg1, BPg2};
  const long total = (long)nNodes * H;
  for (int l = 0; l < 3; ++l) {
    gcn_gemm_kernel<<<(nNodes + 127) / 128, 256, 0, stream>>>(hA, BPgl[l], hB,
                                                              nNodes);
    zero_kernel<<<(int)((total + 255) / 256), 256, 0, stream>>>(hC, total);
    edge_scatter_kernel<<<(int)((nEdges * 32 + 255) / 256), 256, 0, stream>>>(
        erow, ecol, hB, dinv, hC, nEdges);
    self_bias_kernel<<<(int)((total + 255) / 256), 256, 0, stream>>>(
        hB, dinv, bg[l], hC, total);
    ln_res_kernel<<<(int)(((long)nNodes * 32 + 255) / 256), 256, 0, stream>>>(
        hC, gm[l], bt[l], hA, nNodes);
  }

  zero_kernel<<<1, 128, 0, stream>>>(pooled, 128);
  pool_accum_kernel<<<(nNodes + 63) / 64, 256, 0, stream>>>(hA, pooled, nNodes);
  head_kernel<<<1, 32, 0, stream>>>(pooled, W_out, b_out, (float*)d_out,
                                    nNodes);
}
